// PredRNN_25589415149922
// MI455X (gfx1250) — matrix operand order repaired and checked by reading, not compiled, not run on hardware
//
#include <hip/hip_runtime.h>
#include <hip/hip_bf16.h>
#include <math.h>

typedef __attribute__((ext_vector_type(16))) _Float16 v16h;
typedef __attribute__((ext_vector_type(8)))  float    v8f;

#define BSZ   4
#define HID   64
#define CIN   16
#define TT    20
#define TOUT  19
#define HW    1024      // 32*32
#define XW    36        // staged row width: 32 + 2-col halo each side
#define LN_EPS 1e-5f

constexpr int clog2(int n) { return (n <= 1) ? 0 : 1 + clog2(n >> 1); }

// ---------------------------------------------------------------------------
// Pack f32 conv weights into f16 WMMA A-fragments.
// Reduction order: k = r*Cin + ci  (r = tap 0..KK-1, ci = in-channel) so the
// conv-side decode is shifts/masks and a 16-wide lane-half K-slice has
// constant r with consecutive ci. Zero-padded past Kdim.
// k order this code packs (contiguous per lane half; B reads the same):
//   lane l: M = l&15, half = l>>4
//   h[j] : K = half*16 + j
// Stream layout: wp[(tile*KB + kb)*512 + lane*16 + j]
// ---------------------------------------------------------------------------
__global__ void pack_w_kernel(const float* __restrict__ w, _Float16* __restrict__ wp,
                              int Cin, int KK, int Kdim, int KB) {
  int tile = blockIdx.x;
  int kb   = blockIdx.y;
  int t    = threadIdx.x;          // 0..511
  int lane = t >> 4;               // 0..31
  int j    = t & 15;
  int m    = lane & 15;
  int half = lane >> 4;
  int kk   = half * 16 + j;
  int k    = kb * 32 + kk;
  int co   = tile * 16 + m;
  float v  = 0.f;
  if (k < Kdim) {
    int r  = k / Cin;
    int ci = k - r * Cin;
    v = w[(size_t)co * Kdim + (size_t)ci * KK + r];   // original [Cout][Cin][KK]
  }
  wp[((size_t)(tile * KB + kb)) * 512 + lane * 16 + j] = (_Float16)v;
}

// ---------------------------------------------------------------------------
// LDS staging with x-halo: smem[(row*36 + hx)*CP + ci], hx = xx+2,
// columns {0,1,34,35} zeroed. CP = Cin+16 keeps all 16-half fragments 32B
// aligned; all strides are compile-time constants.
// ---------------------------------------------------------------------------
template <int CinT>
__device__ __forceinline__
void stage_lds(const float* __restrict__ x, _Float16* __restrict__ smem,
               int b, int yrow, int rows, int pad) {
  constexpr int CP = CinT + 16;
  constexpr int sh = clog2(CinT);
  const int total = (rows << 5) << sh;                // rows*32*Cin
  for (int e = threadIdx.x; e < total; e += 256) {
    int xx = e & 31;
    int ci = (e >> 5) & (CinT - 1);
    int dy = e >> (5 + sh);
    int gy = yrow + dy - pad;
    float v = 0.f;
    if (gy >= 0 && gy < 32)
      v = x[(((size_t)b * CinT + ci) * 32 + gy) * 32 + xx];
    smem[(dy * XW + xx + 2) * CP + ci] = (_Float16)v;
  }
  const int htot = (rows << 2) << sh;                 // rows*4*Cin
  for (int e = threadIdx.x; e < htot; e += 256) {
    int ci = e & (CinT - 1);
    int c4 = (e >> sh) & 3;
    int dy = e >> (2 + sh);
    int hx = (c4 < 2) ? c4 : (32 + c4);
    smem[(dy * XW + hx) * CP + ci] = (_Float16)0.f;
  }
}

// B-fragment LDS offset (halfs) — branch-free; all strides constant.
// For KK==25 the tap r is clamped to 24 past Kdim (A is zero-padded there,
// so B is don't-care but must stay in-bounds).
template <int CinT, int KKT>
__device__ __forceinline__
int b_off(int kb, int half, int khFix, int kwFix, int xg) {
  constexpr int CP = CinT + 16;
  constexpr int sh = clog2(CinT);
  int kstart = kb * 32 + (half << 4);
  int kh, kw;
  if (KKT == 25) {
    int r = kstart >> sh;
    r = (r > 24) ? 24 : r;
    kh = r / 5; kw = r - kh * 5;
  } else { kh = khFix; kw = kwFix; }
  return (kh * XW + (xg + kw)) * CP + (kstart & (CinT - 1));
}

// ---------------------------------------------------------------------------
// WMMA implicit-GEMM core: one wave computes co tiles {t0, t0+1} x 16 cols.
// Software-pipelined: block kb+1's A/B loads are issued before block kb's
// WMMAs (prefetch index clamped -> last iteration re-reads, harmless), so
// the matrix pipe overlaps the load latency.
// ---------------------------------------------------------------------------
template <int CinT, int KKT, int KBT>
__device__ __forceinline__
void conv_compute(const _Float16* __restrict__ smem,
                  const _Float16* __restrict__ wp, float* __restrict__ y,
                  int n_co, int t0, int khFix, int kwFix,
                  int b, int yrow, size_t out_bstride,
                  int lane, int xg, int half) {
  if (t0 >= n_co) return;                       // wave-uniform
  const bool has1 = (t0 + 1) < n_co;            // wave-uniform
  const _Float16* wp0 = wp + (size_t)t0 * KBT * 512 + lane * 16;
  const _Float16* wp1 = wp0 + (size_t)KBT * 512;

  v8f acc0 = {}, acc1 = {};
  if (has1) {
    v16h a0 = *(const v16h*)wp0;
    v16h a1 = *(const v16h*)wp1;
    v16h bv = *(const v16h*)&smem[b_off<CinT, KKT>(0, half, khFix, kwFix, xg)];
    for (int kb = 0; kb < KBT; ++kb) {
      v16h ca0 = a0, ca1 = a1, cb = bv;
      int nkb = (kb + 1 < KBT) ? (kb + 1) : (KBT - 1);
      a0 = *(const v16h*)(wp0 + (size_t)nkb * 512);
      a1 = *(const v16h*)(wp1 + (size_t)nkb * 512);
      bv = *(const v16h*)&smem[b_off<CinT, KKT>(nkb, half, khFix, kwFix, xg)];
      acc0 = __builtin_amdgcn_wmma_f32_16x16x32_f16(false, ca0, false, cb,
                                                    (short)0, acc0, false, false);
      acc1 = __builtin_amdgcn_wmma_f32_16x16x32_f16(false, ca1, false, cb,
                                                    (short)0, acc1, false, false);
    }
  } else {
    v16h a0 = *(const v16h*)wp0;
    v16h bv = *(const v16h*)&smem[b_off<CinT, KKT>(0, half, khFix, kwFix, xg)];
    for (int kb = 0; kb < KBT; ++kb) {
      v16h ca0 = a0, cb = bv;
      int nkb = (kb + 1 < KBT) ? (kb + 1) : (KBT - 1);
      a0 = *(const v16h*)(wp0 + (size_t)nkb * 512);
      bv = *(const v16h*)&smem[b_off<CinT, KKT>(nkb, half, khFix, kwFix, xg)];
      acc0 = __builtin_amdgcn_wmma_f32_16x16x32_f16(false, ca0, false, cb,
                                                    (short)0, acc0, false, false);
    }
  }

  // D layout: VGPR r -> M = half*8 + r, N = xg
  size_t ybase = (size_t)b * out_bstride + (size_t)yrow * 32 + xg;
  int co0 = t0 * 16 + half * 8;
#pragma unroll
  for (int r2 = 0; r2 < 8; ++r2)
    y[ybase + (size_t)(co0 + r2) * HW] = acc0[r2];
  if (has1) {
#pragma unroll
    for (int r2 = 0; r2 < 8; ++r2)
      y[ybase + (size_t)(co0 + 16 + r2) * HW] = acc1[r2];
  }
}

// ---------------------------------------------------------------------------
// Fused gate convolutions: xcat (448ch), hcat (256ch), mcat (192ch), all 5x5.
// blockIdx.x: [0,4) -> x segment, [4,6) -> h segment, [6,8) -> m segment.
// Template on the x-segment's Cin/KB (16/13 for layer 0, 64/50 otherwise) so
// every stride in the hot loop is a literal. grid = (8, 32, B).
// ---------------------------------------------------------------------------
template <int CINX, int KBX>
__global__ __launch_bounds__(256)
void gateconv_kernel(const float* __restrict__ xin, const float* __restrict__ hin,
                     const float* __restrict__ min_,
                     const _Float16* __restrict__ wpx, const _Float16* __restrict__ wph,
                     const _Float16* __restrict__ wpm,
                     float* __restrict__ xcat, float* __restrict__ hcat,
                     float* __restrict__ mcat) {
  __shared__ _Float16 smem[5 * XW * 80];  // 5 rows * 36 * (64+16) = 28.1 KB max
  const int b = blockIdx.z, yrow = blockIdx.y, bx = blockIdx.x;
  const int wid  = threadIdx.x >> 5;
  const int lane = threadIdx.x & 31;
  const int xg   = ((wid & 1) << 4) + (lane & 15);
  const int half = lane >> 4;

  if (bx < 4) {
    stage_lds<CINX>(xin, smem, b, yrow, 5, 2);
    __syncthreads();
    conv_compute<CINX, 25, KBX>(smem, wpx, xcat, 28, (bx * 4 + (wid >> 1)) * 2,
                                0, 0, b, yrow, (size_t)448 * HW, lane, xg, half);
  } else if (bx < 6) {
    stage_lds<64>(hin, smem, b, yrow, 5, 2);
    __syncthreads();
    conv_compute<64, 25, 50>(smem, wph, hcat, 16, ((bx - 4) * 4 + (wid >> 1)) * 2,
                             0, 0, b, yrow, (size_t)256 * HW, lane, xg, half);
  } else {
    stage_lds<64>(min_, smem, b, yrow, 5, 2);
    __syncthreads();
    conv_compute<64, 25, 50>(smem, wpm, mcat, 12, ((bx - 6) * 4 + (wid >> 1)) * 2,
                             0, 0, b, yrow, (size_t)192 * HW, lane, xg, half);
  }
}

// ---------------------------------------------------------------------------
// Fused Wo (5x5) + Wl (1x1) convolutions over the shared mem=[c|m] tensor.
// Stage the Cin=128 slab once; wave pairs 0-1 -> Wo, pairs 2-3 -> Wl (center
// tap kh=2, kw=2 of the pad-2 halo slab). grid = (1, 32, B).
// ---------------------------------------------------------------------------
__global__ __launch_bounds__(256)
void convoL_kernel(const float* __restrict__ mem,
                   const _Float16* __restrict__ wpo, const _Float16* __restrict__ wpl,
                   float* __restrict__ woo, float* __restrict__ wlo) {
  __shared__ _Float16 smem[5 * XW * 144];   // 5 rows * 36 * (128+16) = 50.6 KB
  const int b = blockIdx.z, yrow = blockIdx.y;
  stage_lds<128>(mem, smem, b, yrow, 5, 2);
  __syncthreads();

  const int wid  = threadIdx.x >> 5;
  const int lane = threadIdx.x & 31;
  const int pair = wid >> 1;
  const int xg   = ((wid & 1) << 4) + (lane & 15);
  const int half = lane >> 4;
  if (pair < 2)
    conv_compute<128, 25, 100>(smem, wpo, woo, 4, pair * 2, 0, 0,
                               b, yrow, (size_t)64 * HW, lane, xg, half);
  else
    conv_compute<128, 1, 4>(smem, wpl, wlo, 4, (pair - 2) * 2, 2, 2,
                            b, yrow, (size_t)64 * HW, lane, xg, half);
}

// ---------------------------------------------------------------------------
// Standalone 1x1 conv for Wout: 64ch -> 16ch. grid = (1, 32, B).
// Single staged row (kh=0); kw=2 hits the halo slab's interior column.
// ---------------------------------------------------------------------------
__global__ __launch_bounds__(256)
void convout_kernel(const float* __restrict__ x, const _Float16* __restrict__ wp,
                    float* __restrict__ y, size_t bstride) {
  __shared__ _Float16 smem[XW * 80];        // 1 row * 36 * (64+16) = 5.6 KB
  const int b = blockIdx.z, yrow = blockIdx.y;
  stage_lds<64>(x, smem, b, yrow, 1, 0);
  __syncthreads();

  const int wid  = threadIdx.x >> 5;
  const int lane = threadIdx.x & 31;
  const int t0   = (wid >> 1) * 2;
  const int xg   = ((wid & 1) << 4) + (lane & 15);
  const int half = lane >> 4;
  conv_compute<64, 1, 2>(smem, wp, y, 1, t0, 0, 2,
                         b, yrow, bstride, lane, xg, half);
}

// ---------------------------------------------------------------------------
// LayerNorm statistics: shared reduction body
// ---------------------------------------------------------------------------
__device__ __forceinline__
void ln_body(const float* __restrict__ x, float* __restrict__ stats, int npb) {
  __shared__ float ssum[256];
  __shared__ float ssq[256];
  int b = blockIdx.x;
  const float* p = x + (size_t)b * npb;
  float s = 0.f, q = 0.f;
  for (int i = threadIdx.x; i < npb; i += 256) {
    float v = p[i];
    s += v; q += v * v;
  }
  ssum[threadIdx.x] = s; ssq[threadIdx.x] = q;
  __syncthreads();
  for (int off = 128; off > 0; off >>= 1) {
    if (threadIdx.x < off) {
      ssum[threadIdx.x] += ssum[threadIdx.x + off];
      ssq[threadIdx.x]  += ssq[threadIdx.x + off];
    }
    __syncthreads();
  }
  if (threadIdx.x == 0) {
    float mu  = ssum[0] / (float)npb;
    float var = ssq[0] / (float)npb - mu * mu;
    stats[b * 2 + 0] = mu;
    stats[b * 2 + 1] = rsqrtf(var + LN_EPS);
  }
}

// grid = (B, 3): one launch computes stats for xcat, hcat, mcat
__global__ void ln3_kernel(const float* __restrict__ x0, const float* __restrict__ x1,
                           const float* __restrict__ x2,
                           float* __restrict__ s0, float* __restrict__ s1,
                           float* __restrict__ s2, int n0, int n1, int n2) {
  const float* x; float* st; int npb;
  if (blockIdx.y == 0)      { x = x0; st = s0; npb = n0; }
  else if (blockIdx.y == 1) { x = x1; st = s1; npb = n1; }
  else                      { x = x2; st = s2; npb = n2; }
  ln_body(x, st, npb);
}

__global__ void ln_stats_kernel(const float* __restrict__ x, float* __restrict__ stats,
                                int npb) {
  ln_body(x, stats, npb);
}

__device__ __forceinline__ float sigmf(float z) { return 1.f / (1.f + expf(-z)); }

// ---------------------------------------------------------------------------
// ST-LSTM gate math: updates c (per layer), m (shared), builds mem=[c|m]
// Index space: B * HID * HW = 262144
// ---------------------------------------------------------------------------
__global__ void gates_kernel(const float* __restrict__ xcat, const float* __restrict__ hcat,
                             const float* __restrict__ mcat,
                             const float* __restrict__ sx, const float* __restrict__ sh,
                             const float* __restrict__ sm,
                             float* __restrict__ c, float* __restrict__ m,
                             float* __restrict__ mem) {
  int tid = blockIdx.x * blockDim.x + threadIdx.x;
  int b   = tid >> 16;          // HID*HW = 65536
  int rem = tid & 65535;        // ch*1024 + p
  float mux = sx[2*b], rsx = sx[2*b+1];
  float muh = sh[2*b], rsh = sh[2*b+1];
  float mum = sm[2*b], rsm = sm[2*b+1];
  size_t xb = (size_t)b * 448 * HW;
  size_t hb = (size_t)b * 256 * HW;
  size_t mb = (size_t)b * 192 * HW;

  float ix  = (xcat[xb + 0*65536 + rem] - mux) * rsx;
  float fx  = (xcat[xb + 1*65536 + rem] - mux) * rsx;
  float gx  = (xcat[xb + 2*65536 + rem] - mux) * rsx;
  float ixp = (xcat[xb + 3*65536 + rem] - mux) * rsx;
  float fxp = (xcat[xb + 4*65536 + rem] - mux) * rsx;
  float gxp = (xcat[xb + 5*65536 + rem] - mux) * rsx;
  float ih  = (hcat[hb + 0*65536 + rem] - muh) * rsh;
  float fh  = (hcat[hb + 1*65536 + rem] - muh) * rsh;
  float gh  = (hcat[hb + 2*65536 + rem] - muh) * rsh;
  float im  = (mcat[mb + 0*65536 + rem] - mum) * rsm;
  float fm  = (mcat[mb + 1*65536 + rem] - mum) * rsm;
  float gm  = (mcat[mb + 2*65536 + rem] - mum) * rsm;

  float i = sigmf(ix + ih);
  float f = sigmf(fx + fh + 1.0f);
  float g = tanhf(gx + gh);
  float c_new = f * c[tid] + i * g;

  float ip = sigmf(ixp + im);
  float fp = sigmf(fxp + fm + 1.0f);
  float gp = tanhf(gxp + gm);
  float m_new = fp * m[tid] + ip * gp;

  c[tid] = c_new;
  m[tid] = m_new;
  size_t memb = (size_t)b * 131072;
  mem[memb + rem]         = c_new;
  mem[memb + 65536 + rem] = m_new;
}

// ---------------------------------------------------------------------------
// h_new = sigmoid(ox + oh + LN(conv(mem,Wo))) * tanh(conv_last(mem))
// ---------------------------------------------------------------------------
__global__ void hcomb_kernel(const float* __restrict__ xcat, const float* __restrict__ hcat,
                             const float* __restrict__ wo, const float* __restrict__ wl,
                             const float* __restrict__ sx, const float* __restrict__ sh,
                             const float* __restrict__ so, float* __restrict__ h) {
  int tid = blockIdx.x * blockDim.x + threadIdx.x;
  int b   = tid >> 16;
  int rem = tid & 65535;
  float mux = sx[2*b], rsx = sx[2*b+1];
  float muh = sh[2*b], rsh = sh[2*b+1];
  float muo = so[2*b], rso = so[2*b+1];
  size_t xb = (size_t)b * 448 * HW;
  size_t hb = (size_t)b * 256 * HW;
  float ox = (xcat[xb + 6*65536 + rem] - mux) * rsx;
  float oh = (hcat[hb + 3*65536 + rem] - muh) * rsh;
  float on = (wo[tid] - muo) * rso;
  float o  = sigmf(ox + oh + on);
  h[tid]   = o * tanhf(wl[tid]);
}

// ---------------------------------------------------------------------------
// Scheduled-sampling blend: x_t = mask*frame + (1-mask)*prev_out  (or frame)
// ---------------------------------------------------------------------------
__global__ void blend_kernel(const float* __restrict__ xp, const float* __restrict__ mp,
                             const float* __restrict__ prev, float* __restrict__ xt,
                             int s) {
  int tid = blockIdx.x * blockDim.x + threadIdx.x;   // B*CIN*HW = 65536
  int b   = tid >> 14;          // CIN*HW = 16384
  int rem = tid & 16383;        // c*1024 + p
  int p   = rem & 1023;
  float frame = xp[((size_t)(b * TT + s)) * 16384 + rem];
  float v = frame;
  if (mp != nullptr) {
    float mk = mp[((size_t)(b * (TT - 2) + (s - 1))) * 1024 + p];
    float pv = prev[(size_t)b * ((size_t)TOUT * 16384) + (size_t)(s - 1) * 16384 + rem];
    v = mk * frame + (1.f - mk) * pv;
  }
  xt[tid] = v;
}

// ---------------------------------------------------------------------------
extern "C" void kernel_launch(void* const* d_in, const int* in_sizes, int n_in,
                              void* d_out, int out_size, void* d_ws, size_t ws_size,
                              hipStream_t stream) {
  (void)in_sizes; (void)n_in; (void)out_size;
  const float* x_patched = (const float*)d_in[0];  // [4,20,16,32,32]
  const float* mask      = (const float*)d_in[1];  // [4,18,1,32,32]
  const float* Wx0       = (const float*)d_in[2];  // [448,16,5,5]
  const float* Wxr       = (const float*)d_in[3];  // [3,448,64,5,5]
  const float* Wh        = (const float*)d_in[4];  // [4,256,64,5,5]
  const float* Wm        = (const float*)d_in[5];  // [4,192,64,5,5]
  const float* Wo        = (const float*)d_in[6];  // [4,64,128,5,5]
  const float* Wl        = (const float*)d_in[7];  // [4,64,128,1,1]
  const float* Wout      = (const float*)d_in[8];  // [16,64,1,1]
  float* out = (float*)d_out;                      // [4,19,16,32,32]

  char* ws = (char*)d_ws;
  size_t off = 0;
  auto carve = [&](size_t bytes) -> void* {
    off = (off + 255) & ~(size_t)255;
    void* p = ws + off;
    off += bytes;
    return p;
  };

  // --- packed f16 weights ---
  _Float16* wpX0 = (_Float16*)carve((size_t)28 * 13  * 512 * 2);
  _Float16* wpXr[3]; for (int l = 0; l < 3; ++l) wpXr[l] = (_Float16*)carve((size_t)28 * 50  * 512 * 2);
  _Float16* wpH[4];  for (int l = 0; l < 4; ++l) wpH[l]  = (_Float16*)carve((size_t)16 * 50  * 512 * 2);
  _Float16* wpM[4];  for (int l = 0; l < 4; ++l) wpM[l]  = (_Float16*)carve((size_t)12 * 50  * 512 * 2);
  _Float16* wpO[4];  for (int l = 0; l < 4; ++l) wpO[l]  = (_Float16*)carve((size_t)4  * 100 * 512 * 2);
  _Float16* wpL[4];  for (int l = 0; l < 4; ++l) wpL[l]  = (_Float16*)carve((size_t)4  * 4   * 512 * 2);
  _Float16* wpOutp = (_Float16*)carve((size_t)1 * 2 * 512 * 2);
  (void)carve(4096);   // slack so pipelined prefetch over-reads stay in-bounds

  // --- activations (f32) ---
  float* xcat = (float*)carve((size_t)BSZ * 448 * HW * 4);
  float* hcat = (float*)carve((size_t)BSZ * 256 * HW * 4);
  float* mcat = (float*)carve((size_t)BSZ * 192 * HW * 4);
  float* wo_o = (float*)carve((size_t)BSZ * 64  * HW * 4);
  float* wl_o = (float*)carve((size_t)BSZ * 64  * HW * 4);
  float* mem  = (float*)carve((size_t)BSZ * 128 * HW * 4);
  float* hbuf = (float*)carve((size_t)4 * BSZ * 64 * HW * 4);
  float* cbuf = (float*)carve((size_t)4 * BSZ * 64 * HW * 4);
  float* mbuf = (float*)carve((size_t)BSZ * 64 * HW * 4);
  float* x_t  = (float*)carve((size_t)BSZ * CIN * HW * 4);
  float* statsX = (float*)carve(BSZ * 2 * 4);
  float* statsH = (float*)carve(BSZ * 2 * 4);
  float* statsM = (float*)carve(BSZ * 2 * 4);
  float* statsO = (float*)carve(BSZ * 2 * 4);
  if (off > ws_size) return;   // workspace too small — bail safely

  // --- pack weights (recomputed every call; deterministic) ---
  auto pack = [&](const float* w, _Float16* wp, int Cout, int Cin, int KK, int KB) {
    pack_w_kernel<<<dim3(Cout / 16, KB), 512, 0, stream>>>(w, wp, Cin, KK, Cin * KK, KB);
  };
  pack(Wx0, wpX0, 448, 16, 25, 13);
  for (int l = 0; l < 3; ++l) pack(Wxr + (size_t)l * 448 * 1600, wpXr[l], 448, 64, 25, 50);
  for (int l = 0; l < 4; ++l) pack(Wh  + (size_t)l * 256 * 1600, wpH[l],  256, 64, 25, 50);
  for (int l = 0; l < 4; ++l) pack(Wm  + (size_t)l * 192 * 1600, wpM[l],  192, 64, 25, 50);
  for (int l = 0; l < 4; ++l) pack(Wo  + (size_t)l * 64  * 3200, wpO[l],  64, 128, 25, 100);
  for (int l = 0; l < 4; ++l) pack(Wl  + (size_t)l * 64  * 128,  wpL[l],  64, 128, 1, 4);
  pack(Wout, wpOutp, 16, 64, 1, 2);

  // --- zero recurrent state each call ---
  hipMemsetAsync(hbuf, 0, (size_t)4 * BSZ * 64 * HW * 4, stream);
  hipMemsetAsync(cbuf, 0, (size_t)4 * BSZ * 64 * HW * 4, stream);
  hipMemsetAsync(mbuf, 0, (size_t)BSZ * 64 * HW * 4, stream);

  const size_t LSTRIDE = (size_t)BSZ * 64 * HW;   // per-layer h/c stride

  for (int s = 0; s < TOUT; ++s) {
    blend_kernel<<<65536 / 256, 256, 0, stream>>>(
        x_patched, s == 0 ? nullptr : mask, s == 0 ? nullptr : out, x_t, s);

    for (int l = 0; l < 4; ++l) {
      const float* xin = (l == 0) ? x_t : (hbuf + (size_t)(l - 1) * LSTRIDE);

      // all three 5x5 gate convs in one launch (x/h/m segments)
      if (l == 0)
        gateconv_kernel<16, 13><<<dim3(8, 32, BSZ), 256, 0, stream>>>(
            xin, hbuf + l * LSTRIDE, mbuf, wpX0, wpH[l], wpM[l],
            xcat, hcat, mcat);
      else
        gateconv_kernel<64, 50><<<dim3(8, 32, BSZ), 256, 0, stream>>>(
            xin, hbuf + l * LSTRIDE, mbuf, wpXr[l - 1], wpH[l], wpM[l],
            xcat, hcat, mcat);

      // all three LN stats in one launch
      ln3_kernel<<<dim3(BSZ, 3), 256, 0, stream>>>(
          xcat, hcat, mcat, statsX, statsH, statsM,
          448 * HW, 256 * HW, 192 * HW);

      gates_kernel<<<1024, 256, 0, stream>>>(xcat, hcat, mcat, statsX, statsH, statsM,
                                             cbuf + l * LSTRIDE, mbuf, mem);

      // Wo (5x5) + Wl (1x1) on the shared mem tensor, single staging pass
      convoL_kernel<<<dim3(1, 32, BSZ), 256, 0, stream>>>(mem, wpO[l], wpL[l],
                                                          wo_o, wl_o);
      ln_stats_kernel<<<BSZ, 256, 0, stream>>>(wo_o, statsO, 64 * HW);

      hcomb_kernel<<<1024, 256, 0, stream>>>(xcat, hcat, wo_o, wl_o,
                                             statsX, statsH, statsO,
                                             hbuf + l * LSTRIDE);
    }

    // output 1x1 conv: h[3] (64ch) -> 16ch, written straight into d_out slice s
    convout_kernel<<<dim3(1, 32, BSZ), 256, 0, stream>>>(
        hbuf + 3 * LSTRIDE, wpOutp, out + (size_t)s * 16384,
        (size_t)TOUT * 16384);
  }
}
